// Encoder_35828617183294
// MI455X (gfx1250) — matrix operand order repaired and checked by reading, not compiled, not run on hardware
//
#include <hip/hip_runtime.h>

typedef unsigned short u16;
typedef __attribute__((ext_vector_type(16))) __bf16 v16bf;
typedef __attribute__((ext_vector_type(8)))  float  v8f;

#define Bdim 128
#define Tdim 256
#define Hdim 512
#define H3   1536
#define Mdim (Bdim * Tdim)   // 32768
#define KSTEP 64

// ---- CDNA5 async global->LDS copy (guarded; falls back to load+ds_store) ----
#if defined(__has_builtin)
#if __has_builtin(__builtin_amdgcn_global_load_async_to_lds_b128)
#define HAVE_ASYNC_LDS 1
#endif
#if __has_builtin(__builtin_amdgcn_s_wait_asynccnt)
#define HAVE_WAIT_ASYNC 1
#endif
#endif
#ifndef HAVE_ASYNC_LDS
#define HAVE_ASYNC_LDS 0
#endif
#ifndef HAVE_WAIT_ASYNC
#define HAVE_WAIT_ASYNC 0
#endif

// builtin expects GCC-vector int4 pointers: AS(1) source, AS(3) LDS dest
typedef int vg4i __attribute__((vector_size(16)));
typedef __attribute__((address_space(1))) vg4i* gp4i;
typedef __attribute__((address_space(3))) vg4i* lp4i;

__device__ __forceinline__ void cp16(u16* dst_lds, const u16* src_global) {
#if HAVE_ASYNC_LDS
    __builtin_amdgcn_global_load_async_to_lds_b128((gp4i)src_global, (lp4i)dst_lds, 0, 0);
#else
    *(uint4*)dst_lds = *(const uint4*)src_global;
#endif
}

__device__ __forceinline__ void stage_wait() {
#if HAVE_ASYNC_LDS
#if HAVE_WAIT_ASYNC
    __builtin_amdgcn_s_wait_asynccnt(0);
#else
    asm volatile("s_wait_asynccnt 0x0" ::: "memory");
#endif
#endif
}

union Frag {
    v16bf v;
    uint4 q[2];
};

__device__ __forceinline__ u16 f2bf(float f) {
    unsigned u = __float_as_uint(f);
    u += 0x7FFFu + ((u >> 16) & 1u);   // round-to-nearest-even
    return (u16)(u >> 16);
}

__device__ __forceinline__ float sigm(float x) {
    return 1.0f / (1.0f + __expf(-x));
}

// ---------------------------------------------------------------------------
// fp32 -> bf16 convert (grid-stride)
// ---------------------------------------------------------------------------
__global__ void k_cvt_bf16(const float* __restrict__ src, u16* __restrict__ dst, int n) {
    int i = blockIdx.x * blockDim.x + threadIdx.x;
    int stride = gridDim.x * blockDim.x;
    for (; i < n; i += stride) dst[i] = f2bf(src[i]);
}

// ---------------------------------------------------------------------------
// weight transpose + convert: [L][512][1536] f32 -> [L][1536][512] bf16 (N-major)
// ---------------------------------------------------------------------------
__global__ void k_wT_bf16(const float* __restrict__ w, u16* __restrict__ wt, int n) {
    int i = blockIdx.x * blockDim.x + threadIdx.x;
    int stride = gridDim.x * blockDim.x;
    for (; i < n; i += stride) {
        int l = i / (Hdim * H3);
        int rem = i - l * (Hdim * H3);
        int k = rem / H3;
        int c = rem - k * H3;
        wt[(size_t)l * H3 * Hdim + (size_t)c * Hdim + k] = f2bf(w[i]);
    }
}

// ---------------------------------------------------------------------------
// Input projection: XP[32768,1536] = X[32768,512](bf16) @ Wk(bf16, N-major) + b0
// 128x64 workgroup tile, 8 waves, K staged 64-wide via async global->LDS copies.
// Per stage/wave: fragments loaded first (batched ds_loads), then 8 WMMAs.
// ---------------------------------------------------------------------------
__global__ __launch_bounds__(256) void k_xproj(const u16* __restrict__ X,
                                               const u16* __restrict__ WT,
                                               const float* __restrict__ bias0,
                                               float* __restrict__ XP) {
    __shared__ __align__(16) u16 lA[128 * KSTEP];  // 16 KB
    __shared__ __align__(16) u16 lB[64 * KSTEP];   // 8 KB

    const int tid  = threadIdx.x;
    const int w    = tid >> 5;
    const int lane = tid & 31;
    const int half = lane >> 4;
    const int ln   = lane & 15;
    const int m0   = (blockIdx.x / 24) * 128;
    const int n0   = (blockIdx.x % 24) * 64;

    v8f acc[4] = {};

    for (int kk = 0; kk < 512; kk += KSTEP) {
        // stage A tile (128 x 64): 1024 b128 slots, 4 per thread
#pragma unroll
        for (int i = 0; i < 4; ++i) {
            int slot = tid + 256 * i;
            int row  = slot >> 3;
            int cc   = (slot & 7) * 8;
            cp16(&lA[row * KSTEP + cc], &X[(size_t)(m0 + row) * 512 + kk + cc]);
        }
        // stage B tile (64 x 64, N-major): 512 slots, 2 per thread
#pragma unroll
        for (int i = 0; i < 2; ++i) {
            int slot = tid + 256 * i;
            int row  = slot >> 3;
            int cc   = (slot & 7) * 8;
            cp16(&lB[row * KSTEP + cc], &WT[(size_t)(n0 + row) * 512 + kk + cc]);
        }
        {   // prefetch next K-slice of A toward L2
            int kn = (kk + KSTEP < 512) ? kk + KSTEP : kk;
            __builtin_prefetch(&X[(size_t)(m0 + (tid >> 3)) * 512 + kn], 0, 1);
        }
        stage_wait();
        __syncthreads();

#pragma unroll
        for (int ks = 0; ks < KSTEP; ks += 32) {
            // A fragment: lane m = ln, contiguous k = half*16 .. half*16+15
            Frag a;
            const u16* Ar = &lA[(w * 16 + ln) * KSTEP + ks];
            a.q[0] = *(const uint4*)(Ar + half * 16);
            a.q[1] = *(const uint4*)(Ar + half * 16 + 8);
            // all 4 B fragments first (batched ds_loads, single wait) ...
            Frag b0, b1, b2, b3;
            const u16* Br0 = &lB[(0 * 16 + ln) * KSTEP + ks + half * 16];
            const u16* Br1 = &lB[(1 * 16 + ln) * KSTEP + ks + half * 16];
            const u16* Br2 = &lB[(2 * 16 + ln) * KSTEP + ks + half * 16];
            const u16* Br3 = &lB[(3 * 16 + ln) * KSTEP + ks + half * 16];
            b0.q[0] = *(const uint4*)(Br0); b0.q[1] = *(const uint4*)(Br0 + 8);
            b1.q[0] = *(const uint4*)(Br1); b1.q[1] = *(const uint4*)(Br1 + 8);
            b2.q[0] = *(const uint4*)(Br2); b2.q[1] = *(const uint4*)(Br2 + 8);
            b3.q[0] = *(const uint4*)(Br3); b3.q[1] = *(const uint4*)(Br3 + 8);
            // ... then 4 independent WMMA chains back-to-back
            acc[0] = __builtin_amdgcn_wmma_f32_16x16x32_bf16(false, a.v, false, b0.v, (short)0, acc[0], false, false);
            acc[1] = __builtin_amdgcn_wmma_f32_16x16x32_bf16(false, a.v, false, b1.v, (short)0, acc[1], false, false);
            acc[2] = __builtin_amdgcn_wmma_f32_16x16x32_bf16(false, a.v, false, b2.v, (short)0, acc[2], false, false);
            acc[3] = __builtin_amdgcn_wmma_f32_16x16x32_bf16(false, a.v, false, b3.v, (short)0, acc[3], false, false);
        }
        __syncthreads();
    }

    // C/D layout: row = v + 8*half, col = ln
#pragma unroll
    for (int nt = 0; nt < 4; ++nt)
#pragma unroll
        for (int v = 0; v < 8; ++v) {
            int row = m0 + w * 16 + v + 8 * half;
            int col = n0 + nt * 16 + ln;
            XP[(size_t)row * H3 + col] = acc[nt][v] + bias0[col];
        }
}

// ---------------------------------------------------------------------------
// Persistent GRU recurrence. 64 WGs = 32 column-groups (16 cols of H each) x
// 2 row-halves (64 batch rows each), 128 threads (4 waves). Recurrent weight
// slice (3 gates x 16 cols x 512 K = 48 KB bf16) lives in LDS for all T steps.
// h ping-pongs in global (f32 for gate math, bf16 for WMMA A operand, L2-hot).
// Grid-wide sense barrier between steps.
// ---------------------------------------------------------------------------
#define RWG_COLS 32
#define RGRID    64
#define RTHREADS 128

__global__ __launch_bounds__(RTHREADS) void k_gru_rec(
    const float* __restrict__ XP,    // [B*T, 1536] f32 (row = b*T + t)
    const u16* __restrict__ WuT,     // [1536, 512] bf16 (N-major)
    const float* __restrict__ br,    // [1536] recurrent bias
    float* __restrict__ hf0, float* __restrict__ hf1,   // h f32 ping-pong
    u16* __restrict__ hb0, u16* __restrict__ hb1,       // h bf16 ping-pong
    u16* __restrict__ Ybf,           // [B*T, 512] bf16 (next layer input)
    float* __restrict__ Yout,        // d_out for last layer, else nullptr
    unsigned* cnt, unsigned* gen) {
    __shared__ __align__(16) u16 lW[3 * 16 * 512];   // 48 KB

    const int tid  = threadIdx.x;
    const int w    = tid >> 5;
    const int lane = tid & 31;
    const int half = lane >> 4;
    const int ln   = lane & 15;
    const int g       = blockIdx.x % RWG_COLS;
    const int rowBase = (blockIdx.x / RWG_COLS) * 64;
    const int j0      = g * 16;

    // preload recurrent-weight slice: rows {j0..j0+15} of each gate, all K
    for (int i = tid * 8; i < 3 * 16 * 512; i += RTHREADS * 8) {
        int gate = i >> 13;           // /(16*512)
        int rem  = i & 8191;
        int rrow = rem >> 9;          // /512
        int kc   = rem & 511;
        cp16(&lW[i], &WuT[(size_t)(gate * 512 + j0 + rrow) * 512 + kc]);
    }
    stage_wait();
    __syncthreads();

    const int bA = rowBase + w * 16 + ln;   // A-fragment batch row

    for (int t = 0; t < Tdim; ++t) {
        const float* hc = (t & 1) ? hf1 : hf0;
        float*       hn = (t & 1) ? hf0 : hf1;
        const u16*  hbc = (t & 1) ? hb1 : hb0;
        u16*        hbn = (t & 1) ? hb0 : hb1;

        v8f acc[3] = {};
#pragma unroll 2
        for (int kk = 0; kk < 512; kk += 32) {
            Frag a;   // A = h (bf16), direct from global (L2-hot)
            const u16* Ar = &hbc[(size_t)bA * 512 + kk];
            a.q[0] = *(const uint4*)(Ar + half * 16);
            a.q[1] = *(const uint4*)(Ar + half * 16 + 8);
            // batch the 3 gate B-fragments, then 3 back-to-back WMMA
            Frag b0, b1, b2;
            const u16* Br0 = &lW[(0 * 16 + ln) * 512 + kk + half * 16];
            const u16* Br1 = &lW[(1 * 16 + ln) * 512 + kk + half * 16];
            const u16* Br2 = &lW[(2 * 16 + ln) * 512 + kk + half * 16];
            b0.q[0] = *(const uint4*)(Br0); b0.q[1] = *(const uint4*)(Br0 + 8);
            b1.q[0] = *(const uint4*)(Br1); b1.q[1] = *(const uint4*)(Br1 + 8);
            b2.q[0] = *(const uint4*)(Br2); b2.q[1] = *(const uint4*)(Br2 + 8);
            acc[0] = __builtin_amdgcn_wmma_f32_16x16x32_bf16(false, a.v, false, b0.v, (short)0, acc[0], false, false);
            acc[1] = __builtin_amdgcn_wmma_f32_16x16x32_bf16(false, a.v, false, b1.v, (short)0, acc[1], false, false);
            acc[2] = __builtin_amdgcn_wmma_f32_16x16x32_bf16(false, a.v, false, b2.v, (short)0, acc[2], false, false);
        }

        // gates: z, r, h (Keras reset_after: r multiplies (rec_h + br_h))
#pragma unroll
        for (int v = 0; v < 8; ++v) {
            const int b = rowBase + w * 16 + v + 8 * half;
            const int j = j0 + ln;
            const float* xr = &XP[((size_t)b * Tdim + t) * H3];
            float z  = sigm(xr[j] + acc[0][v] + br[j]);
            float rr = sigm(xr[512 + j] + acc[1][v] + br[512 + j]);
            float hh = tanhf(xr[1024 + j] + rr * (acc[2][v] + br[1024 + j]));
            float ho = hc[(size_t)b * 512 + j];
            float o  = z * ho + (1.0f - z) * hh;
            hn[(size_t)b * 512 + j] = o;
            u16 ob = f2bf(o);
            hbn[(size_t)b * 512 + j] = ob;
            Ybf[((size_t)b * Tdim + t) * 512 + j] = ob;
            if (Yout) Yout[(size_t)b * Tdim * 512 + (size_t)t * 512 + j] = o;
        }

        // grid-wide sense barrier between recurrence steps
        __threadfence();
        __syncthreads();
        if (tid == 0) {
            volatile unsigned* vgen = gen;
            unsigned old = *vgen;
            if (atomicAdd(cnt, 1u) == (unsigned)(gridDim.x - 1)) {
                *(volatile unsigned*)cnt = 0u;
                __threadfence();
                *vgen = old + 1u;
            } else {
                while (*vgen == old) __builtin_amdgcn_s_sleep(2);
            }
        }
        __syncthreads();
    }
}

// ---------------------------------------------------------------------------
// Host launcher
// ---------------------------------------------------------------------------
extern "C" void kernel_launch(void* const* d_in, const int* in_sizes, int n_in,
                              void* d_out, int out_size, void* d_ws, size_t ws_size,
                              hipStream_t stream) {
    (void)in_sizes; (void)n_in; (void)out_size; (void)ws_size;
    const float* x  = (const float*)d_in[0];
    const float* Wk = (const float*)d_in[1];
    const float* Wu = (const float*)d_in[2];
    const float* bs = (const float*)d_in[3];
    float* out = (float*)d_out;

    char* ws = (char*)d_ws;
    size_t off = 0;
    auto carve = [&](size_t bytes) -> void* {
        void* p = ws + off;
        off = (off + bytes + 255) & ~(size_t)255;
        return p;
    };
    float* XP   = (float*)carve((size_t)Mdim * H3 * 4);        // 201 MB
    u16*   XbfA = (u16*)carve((size_t)Mdim * Hdim * 2);        // 32 MB
    u16*   XbfB = (u16*)carve((size_t)Mdim * Hdim * 2);        // 32 MB
    u16*   WkT  = (u16*)carve((size_t)3 * H3 * Hdim * 2);      // 4.7 MB
    u16*   WuT  = (u16*)carve((size_t)3 * H3 * Hdim * 2);      // 4.7 MB
    float* hf   = (float*)carve((size_t)2 * Bdim * Hdim * 4);  // 512 KB
    u16*   hb   = (u16*)carve((size_t)2 * Bdim * Hdim * 2);    // 256 KB
    unsigned* sync = (unsigned*)carve(512);
    unsigned* cnt = sync;
    unsigned* gen = sync + 64;

    (void)hipMemsetAsync(sync, 0, 512, stream);
    (void)hipMemsetAsync(hf, 0, (size_t)2 * Bdim * Hdim * 4, stream);  // h0 = 0
    (void)hipMemsetAsync(hb, 0, (size_t)2 * Bdim * Hdim * 2, stream);

    k_cvt_bf16<<<2048, 256, 0, stream>>>(x, XbfA, Mdim * Hdim);
    k_wT_bf16<<<2048, 256, 0, stream>>>(Wk, WkT, 3 * Hdim * H3);
    k_wT_bf16<<<2048, 256, 0, stream>>>(Wu, WuT, 3 * Hdim * H3);

    for (int l = 0; l < 3; ++l) {
        const u16* Xin = (l & 1) ? XbfB : XbfA;   // l=0:A->B, l=1:B->A, l=2:A->B
        u16*       Yb  = (l & 1) ? XbfA : XbfB;
        const float* bias0 = bs + (size_t)l * 2 * H3;
        const float* bias1 = bias0 + H3;
        k_xproj<<<(Mdim / 128) * (H3 / 64), 256, 0, stream>>>(
            Xin, WkT + (size_t)l * H3 * Hdim, bias0, XP);
        float* Yout = (l == 2) ? out : nullptr;
        k_gru_rec<<<RGRID, RTHREADS, 0, stream>>>(
            XP, WuT + (size_t)l * H3 * Hdim, bias1,
            hf, hf + Bdim * Hdim, hb, hb + Bdim * Hdim,
            Yb, Yout, cnt, gen);
    }
    // final state of layer 2 lands in ping-pong buffer 0 (T even)
    (void)hipMemcpyAsync(out + (size_t)Tdim * Bdim * Hdim, hf,
                         (size_t)Bdim * Hdim * 4, hipMemcpyDeviceToDevice, stream);
}
